// ODEfunc_72335839199610
// MI455X (gfx1250) — matrix operand order repaired and checked by reading, not compiled, not run on hardware
//
#include <hip/hip_runtime.h>
#include <hip/hip_bf16.h>

// gfx1250 / CDNA5: wave32, WMMA (no MFMA). GEMM runs as bf16x3 split-precision
// on v_wmma_f32_16x16x32_bf16 with f32 accumulation (~fp32 accuracy), which is
// ~8x the per-instruction FLOPs of the f32 16x16x4 WMMA and keeps the whole
// pipeline HBM-bound (~0.65 GB traffic @ 23.3 TB/s ~ 28us).
//
// Key algebraic cut: the reference computes h = relu(x[:3N] @ W^T + b) but the
// R block (rows 2N..3N) is never consumed (dR = gam * I), so the GEMM only
// runs over the S and I blocks (2N rows).

#define HID 128

typedef __attribute__((ext_vector_type(16))) __bf16 v16bf;
typedef __attribute__((ext_vector_type(8)))  float  v8f;

// ---------------------------------------------------------------------------
// Kernel 1: h = relu(x[:2N] @ W^T + b), bf16x3 split GEMM with WMMA.
// One block = 256 threads = 8 waves; each wave owns a 16-row x 128-col tile.
// W (128x128 f32) is split into hi/lo bf16 planes staged in LDS (64 KB).
// ---------------------------------------------------------------------------
__global__ __launch_bounds__(256) void gemm_relu_kernel(
    const float* __restrict__ x, const float* __restrict__ W,
    const float* __restrict__ bias, float* __restrict__ hout, int M)
{
    __shared__ __bf16 Whi[HID * HID];
    __shared__ __bf16 Wlo[HID * HID];

    // Cooperative load + hi/lo split of W (B-matrix source, L2-resident).
    for (int idx = threadIdx.x; idx < HID * HID; idx += 256) {
        float w = W[idx];
        __bf16 hi = (__bf16)w;
        Whi[idx] = hi;
        Wlo[idx] = (__bf16)(w - (float)hi);
    }
    __syncthreads();

    const int wave    = threadIdx.x >> 5;
    const int lane    = threadIdx.x & 31;
    const int halfsel = lane >> 4;      // 0: lanes 0-15, 1: lanes 16-31
    const int lane16  = lane & 15;
    const int m0      = (blockIdx.x * 8 + wave) * 16;

    // A-matrix row for this lane (ISA 16-bit A 16x32 layout: M = lane%16).
    int arow = m0 + lane16;
    if (arow >= M) arow = M - 1;                 // clamp; stores are guarded
    const float* __restrict__ xrow = x + (size_t)arow * HID;

    v8f acc[8];
    const v8f vzero = {0.f, 0.f, 0.f, 0.f, 0.f, 0.f, 0.f, 0.f};
#pragma unroll
    for (int j = 0; j < 8; ++j) acc[j] = vzero;

#pragma unroll
    for (int kc = 0; kc < HID; kc += 32) {
        // A 16x32 bf16 layout: lanes 0-15 hold K = kc+{0..15},
        // lanes 16-31 hold K = kc+{16..31}, in two runs of 8.
        const int kA0 = kc + halfsel * 16;
        const int kA1 = kc + halfsel * 16 + 8;
        v16bf a_hi, a_lo;
#pragma unroll
        for (int u = 0; u < 8; ++u) {
            float v0 = xrow[kA0 + u];
            __bf16 h0 = (__bf16)v0;
            a_hi[u]     = h0;
            a_lo[u]     = (__bf16)(v0 - (float)h0);
            float v1 = xrow[kA1 + u];
            __bf16 h1 = (__bf16)v1;
            a_hi[8 + u] = h1;
            a_lo[8 + u] = (__bf16)(v1 - (float)h1);
        }

#pragma unroll
        for (int jt = 0; jt < 8; ++jt) {
            // B 32x16 bf16: B[k][nn] = W[nn][k]; lanes 0-15 carry K=kc..kc+15,
            // lanes 16-31 carry K=kc+16..kc+31, column nn = jt*16 + lane16.
            const int bn = jt * 16 + lane16;
            const int bk = kc + halfsel * 16;
            v16bf b_hi, b_lo;
#pragma unroll
            for (int u = 0; u < 16; ++u) {
                b_hi[u] = Whi[bn * HID + bk + u];
                b_lo[u] = Wlo[bn * HID + bk + u];
            }
            // bf16x3: hi*hi + lo*hi + hi*lo, f32 accumulate.
            acc[jt] = __builtin_amdgcn_wmma_f32_16x16x32_bf16(
                false, a_hi, false, b_hi, (short)0, acc[jt], false, false);
            acc[jt] = __builtin_amdgcn_wmma_f32_16x16x32_bf16(
                false, a_lo, false, b_hi, (short)0, acc[jt], false, false);
            acc[jt] = __builtin_amdgcn_wmma_f32_16x16x32_bf16(
                false, a_hi, false, b_lo, (short)0, acc[jt], false, false);
        }
    }

    // Epilogue: C/D layout — VGPR e: lanes 0-15 -> M=e, lanes 16-31 -> M=e+8.
#pragma unroll
    for (int jt = 0; jt < 8; ++jt) {
        const int col = jt * 16 + lane16;
        const float bc = bias[col];
#pragma unroll
        for (int e = 0; e < 8; ++e) {
            const int row = m0 + e + halfsel * 8;
            if (row < M) {
                float v = acc[jt][e] + bc;
                hout[(size_t)row * HID + col] = v > 0.f ? v : 0.f;
            }
        }
    }
}

// ---------------------------------------------------------------------------
// Kernel 2: zero AI accumulator (d_ws is poisoned, and must be re-zeroed
// every call since scatter accumulates).
// ---------------------------------------------------------------------------
__global__ void zero_kernel(float* __restrict__ p, long long n)
{
    long long i = (long long)blockIdx.x * blockDim.x + threadIdx.x;
    long long stride = (long long)gridDim.x * blockDim.x;
    for (; i < n; i += stride) p[i] = 0.f;
}

// ---------------------------------------------------------------------------
// Kernel 3: AI[row] += I[col] per edge. One wave per edge; lane l moves a
// float4 (cols 4l..4l+3). I block (51 MB) is L2-resident; atomics are
// hardware global_atomic_add_f32 (no-return) landing in L2 atomic units.
// ---------------------------------------------------------------------------
__global__ __launch_bounds__(256) void scatter_kernel(
    const int* __restrict__ erow, const int* __restrict__ ecol,
    const float* __restrict__ Ih, float* __restrict__ AI, int nE)
{
    const int wid  = (int)((blockIdx.x * (unsigned)blockDim.x + threadIdx.x) >> 5);
    const int lane = threadIdx.x & 31;
    if (wid >= nE) return;
    const int r = erow[wid];
    const int c = ecol[wid];
    const float4 v = ((const float4*)(Ih + (size_t)c * HID))[lane];
    float* dst = AI + (size_t)r * HID + lane * 4;
    unsafeAtomicAdd(dst + 0, v.x);
    unsafeAtomicAdd(dst + 1, v.y);
    unsafeAtomicAdd(dst + 2, v.z);
    unsafeAtomicAdd(dst + 3, v.w);
}

// ---------------------------------------------------------------------------
// Kernel 4: SIR dynamics + per-row LayerNorm + tail copy. One wave per node,
// 4 cols/lane, wave32 shfl_xor reductions (no LDS, no barriers).
// dR = gam * I, so the R block of h is never needed.
// ---------------------------------------------------------------------------
__global__ __launch_bounds__(256) void finalize_kernel(
    const float* __restrict__ x, const float* __restrict__ hbuf,
    const float* __restrict__ AI, const float* __restrict__ ln_w,
    const float* __restrict__ ln_b, float* __restrict__ out, int n)
{
    const int wid  = (int)((blockIdx.x * (unsigned)blockDim.x + threadIdx.x) >> 5);
    const int lane = threadIdx.x & 31;
    if (wid >= n) return;

    const float* __restrict__ Srow = hbuf + (size_t)wid * HID;
    const float* __restrict__ Irow = hbuf + (size_t)(n + wid) * HID;
    const float* __restrict__ Arow = AI + (size_t)wid * HID;
    const float* __restrict__ tail = x + (size_t)(3 * (size_t)n + wid) * HID;

    const float beta = tail[0];   // broadcast load
    const float gam  = tail[1];

    float dS[4], dI[4], dR[4], tl[4];
#pragma unroll
    for (int u = 0; u < 4; ++u) {
        const int c = lane + 32 * u;
        const float s  = Srow[c];
        const float ii = Irow[c];
        const float ai = Arow[c];
        const float t1 = beta * (ai * s);
        dS[u] = -t1;
        dI[u] = t1 - gam * ii;
        dR[u] = gam * ii;
        tl[u] = tail[c];
    }

    auto ln_store = [&](const float v[4], float* __restrict__ orow) {
        float s = v[0] + v[1] + v[2] + v[3];
#pragma unroll
        for (int off = 16; off > 0; off >>= 1) s += __shfl_xor(s, off, 32);
        const float m = s * (1.f / 128.f);
        float q = 0.f;
#pragma unroll
        for (int u = 0; u < 4; ++u) { const float d = v[u] - m; q += d * d; }
#pragma unroll
        for (int off = 16; off > 0; off >>= 1) q += __shfl_xor(q, off, 32);
        const float inv = rsqrtf(q * (1.f / 128.f) + 1e-5f);
#pragma unroll
        for (int u = 0; u < 4; ++u) {
            const int c = lane + 32 * u;
            orow[c] = (v[u] - m) * inv * ln_w[c] + ln_b[c];
        }
    };

    ln_store(dS, out + (size_t)wid * HID);
    ln_store(dI, out + (size_t)(n + wid) * HID);
    ln_store(dR, out + (size_t)(2 * (size_t)n + wid) * HID);
#pragma unroll
    for (int u = 0; u < 4; ++u) {
        const int c = lane + 32 * u;
        out[(size_t)(3 * (size_t)n + wid) * HID + c] = tl[u];
    }
}

// ---------------------------------------------------------------------------
extern "C" void kernel_launch(void* const* d_in, const int* in_sizes, int n_in,
                              void* d_out, int out_size, void* d_ws, size_t ws_size,
                              hipStream_t stream)
{
    // setup_inputs order: t, x, edge_row, edge_col, W, b, ln_w, ln_b
    const float* x    = (const float*)d_in[1];
    const int*   erow = (const int*)d_in[2];
    const int*   ecol = (const int*)d_in[3];
    const float* W    = (const float*)d_in[4];
    const float* b    = (const float*)d_in[5];
    const float* lnw  = (const float*)d_in[6];
    const float* lnb  = (const float*)d_in[7];
    float* out = (float*)d_out;

    const int n4 = in_sizes[1] / HID;   // 4N rows
    const int n  = n4 / 4;              // N nodes
    const int M  = 2 * n;               // GEMM rows: S and I blocks only
    const int nE = in_sizes[2];         // edges

    float* hbuf = (float*)d_ws;                     // 2N x 128 f32 (S, I)
    float* AI   = hbuf + (size_t)M * HID;           // N  x 128 f32

    // 1) GEMM + ReLU over S and I blocks (R is provably dead).
    const int gemm_blocks = (M + 127) / 128;
    gemm_relu_kernel<<<gemm_blocks, 256, 0, stream>>>(x, W, b, hbuf, M);

    // 2) Zero AI accumulator.
    const long long ai_elems = (long long)n * HID;
    zero_kernel<<<2048, 256, 0, stream>>>(AI, ai_elems);

    // 3) Edge scatter-add (hardware f32 atomics into L2-resident AI).
    const float* Ih = hbuf + (size_t)n * HID;
    scatter_kernel<<<(nE + 7) / 8, 256, 0, stream>>>(erow, ecol, Ih, AI, nE);

    // 4) Dynamics + LayerNorm + tail copy.
    finalize_kernel<<<(n + 7) / 8, 256, 0, stream>>>(x, hbuf, AI, lnw, lnb, out, n);
}
